// MLPPredictor_89043261980805
// MI455X (gfx1250) — matrix operand order repaired and checked by reading, not compiled, not run on hardware
//
#include <hip/hip_runtime.h>

typedef __attribute__((ext_vector_type(16))) _Float16 v16h;
typedef __attribute__((ext_vector_type(8)))  float    v8f;

#define DF 5            // D_FEAT
#define NC 16           // OUT_CLASSES
#define TWODF 10        // 2*D_FEAT (K dimension, zero-padded to 32 for WMMA)

// One wave32 computes one FULL 16-edge tile: D[16x16] = A[16x32] x B[32x16] + bias.
// No bounds checks anywhere: host only launches this over full tiles.
__global__ __launch_bounds__(256) void edge_mlp_wmma_full(
    const float* __restrict__ h,      // [N_NODES, 5]
    const int*   __restrict__ src,    // [E] (offset-adjusted by host)
    const int*   __restrict__ dst,    // [E]
    const float* __restrict__ W,      // [16, 10] row-major
    const float* __restrict__ bias,   // [16]
    float*       __restrict__ out)    // [E, 16]
{
    const int lane = threadIdx.x & 31;                     // wave32
    const int tile = blockIdx.x * 8 + (threadIdx.x >> 5);  // 16 edges per wave
    const int col  = lane & 15;
    const bool hi  = lane >= 16;
    const int  e   = tile * 16 + col;

    // ---- B matrix (32x16 f16): lane<16 holds column N=col, K=0..15 in halves;
    //      lanes 16..31 cover K=16..31 = zero padding.
    v16h b = {};
    if (!hi) {
        #pragma unroll
        for (int k = 0; k < TWODF; ++k)
            b[k] = (_Float16)W[col * TWODF + k];           // B[k][n] = W[n][k]
    }

    // ---- C accumulator preloaded with bias: C[v][lane] = bias[col]
    const float bv = bias[col];
    v8f acc;
    #pragma unroll
    for (int i = 0; i < 8; ++i) acc[i] = bv;

    // ---- A matrix (16x32 f16), zero-padded beyond K=9.
    // lane<16  (M=col): halves[0..9] = K0..9  (h[src][0..4], h[dst][0..4])
    // lane>=16 (M=col): all zero (K16..31 padding)
    v16h a = {};
    if (!hi) {
        const float* hs = h + (size_t)src[e] * DF;
        const float* hd = h + (size_t)dst[e] * DF;
        a[0] = (_Float16)hs[0];
        a[1] = (_Float16)hs[1];
        a[2] = (_Float16)hs[2];
        a[3] = (_Float16)hs[3];
        a[4] = (_Float16)hs[4];
        a[5] = (_Float16)hd[0];
        a[6] = (_Float16)hd[1];
        a[7] = (_Float16)hd[2];
        a[8] = (_Float16)hd[3];   // K=8
        a[9] = (_Float16)hd[4];   // K=9
    }

    // All lanes active (EXEC == all 1s, required for WMMA).
    // emits: v_wmma_f32_16x16x32_f16
    acc = __builtin_amdgcn_wmma_f32_16x16x32_f16(
        /*neg_a=*/false, a, /*neg_b=*/false, b,
        /*c_mod=*/(short)0, acc, /*reuse_a=*/false, /*reuse_b=*/false);

    // ---- Store D: VGPR v of lane L = (M = v + (hi?8:0), N = col).
    // Unconditional: 8 global_store_b32 in one clause, immediate offsets;
    // each store covers two contiguous 64B rows across the wave (coalesced).
    float* o = out + (size_t)tile * (16 * NC) + (hi ? 8 * NC : 0) + col;
    #pragma unroll
    for (int v = 0; v < 8; ++v)
        o[v * NC] = acc[v];
}

// Exact-f32 scalar cleanup for the last E % 16 edges (one thread per (edge, class)).
// Never launched when E is a multiple of 16 (e.g. E = 6,400,000).
__global__ void edge_mlp_tail(
    const float* __restrict__ h,
    const int*   __restrict__ src,
    const int*   __restrict__ dst,
    const float* __restrict__ W,
    const float* __restrict__ bias,
    float*       __restrict__ out,
    int e_start, int n_edges)
{
    const int t = (int)threadIdx.x;                // 256 threads cover 16 edges x 16 classes
    const int e = e_start + t / NC;
    const int c = t % NC;
    if (e >= n_edges) return;

    const float* hs = h + (size_t)src[e] * DF;
    const float* hd = h + (size_t)dst[e] * DF;
    float s = bias[c];
    #pragma unroll
    for (int k = 0; k < DF; ++k) {
        s += hs[k] * W[c * TWODF + k];
        s += hd[k] * W[c * TWODF + DF + k];
    }
    out[(size_t)e * NC + c] = s;
}

extern "C" void kernel_launch(void* const* d_in, const int* in_sizes, int n_in,
                              void* d_out, int out_size, void* d_ws, size_t ws_size,
                              hipStream_t stream) {
    const float* h    = (const float*)d_in[0];
    const int*   src  = (const int*)  d_in[1];
    const int*   dst  = (const int*)  d_in[2];
    const float* W    = (const float*)d_in[3];
    const float* bias = (const float*)d_in[4];
    float*       out  = (float*)d_out;

    const int n_edges    = in_sizes[1];          // E = 6,400,000
    const int full_tiles = n_edges / 16;         // guard-free 16-edge WMMA tiles
    const int rem        = n_edges - full_tiles * 16;

    const int full_blocks = full_tiles / 8;      // 8 wave-tiles per 256-thread block
    const int tile_rem    = full_tiles - full_blocks * 8;

    if (full_blocks > 0)
        edge_mlp_wmma_full<<<full_blocks, 256, 0, stream>>>(h, src, dst, W, bias, out);

    if (tile_rem > 0) {
        const int eoff = full_blocks * 8 * 16;   // edges already covered
        edge_mlp_wmma_full<<<1, tile_rem * 32, 0, stream>>>(
            h, src + eoff, dst + eoff, W, bias, out + (size_t)eoff * NC);
    }

    if (rem > 0)
        edge_mlp_tail<<<1, 256, 0, stream>>>(h, src, dst, W, bias, out,
                                             full_tiles * 16, n_edges);
}
